// layer_34522947125928
// MI455X (gfx1250) — matrix operand order repaired and checked by reading, not compiled, not run on hardware
//
#include <hip/hip_runtime.h>
#include <hip/hip_bf16.h>

// Problem constants (match reference)
#define BB 8
#define TT 128
#define SS 512
#define DD 128

typedef __attribute__((ext_vector_type(16))) _Float16 v16h;
typedef __attribute__((ext_vector_type(8)))  float    v8f;

__device__ __forceinline__ float sigmoidf_(float x) {
    return 1.0f / (1.0f + __expf(-x));
}

// ---------------------------------------------------------------------------
// Pre-swizzle kernels: pack f32 matrices into WMMA f16 A / B lane layouts.
//
// A-layout (16x32 f16 tile, one tile per wave, per ISA 7.12.2):
//   flat index = ((mtile*4 + kblock)*32 + lane)*16 + e
//   m = mtile*16 + (lane&15)
//   k = kblock*32 + (e&7) + ((e&8)?16:0) + 8*(lane>>4)
// B-layout (32x16 f16 tile):
//   flat index = ((kblock*NT + ntile)*32 + lane)*16 + e
//   n = ntile*16 + (lane&15)
//   k = kblock*32 + (e&7) + ((e&8)?16:0) + 8*(lane>>4)
// ---------------------------------------------------------------------------

__global__ void swizzleA_kernel(_Float16* __restrict__ dst,
                                const float* __restrict__ src, int total) {
    int i = blockIdx.x * 256 + threadIdx.x;
    if (i >= total) return;
    int e  = i & 15;
    int l  = (i >> 4) & 31;
    int kb = (i >> 9) & 3;
    int mt = i >> 11;
    int m  = mt * 16 + (l & 15);
    int k  = kb * 32 + (e & 7) + ((e & 8) ? 16 : 0) + 8 * (l >> 4);
    dst[i] = (_Float16)src[m * DD + k];
}

__global__ void swizzleB_kernel(_Float16* __restrict__ dst,
                                const float* __restrict__ s0,
                                const float* __restrict__ s1,
                                const float* __restrict__ s2,
                                const float* __restrict__ s3,
                                const float* __restrict__ s4, int NT) {
    int total = 4 * NT * 32 * 16;
    int i = blockIdx.x * 256 + threadIdx.x;
    if (i >= total) return;
    int e  = i & 15;
    int l  = (i >> 4) & 31;
    int q2 = i >> 9;
    int nt = q2 % NT;
    int kb = q2 / NT;
    int n  = nt * 16 + (l & 15);
    int k  = kb * 32 + (e & 7) + ((e & 8) ? 16 : 0) + 8 * (l >> 4);
    const float* srcs[5] = {s0, s1, s2, s3, s4};
    int g = n >> 7;          // which 128-col block
    int d = n & 127;
    dst[i] = (_Float16)srcs[g][k * DD + d];
}

// bias4, h0/c0 init, q0 = h0 @ Wa
__global__ void misc_prep_kernel(float* __restrict__ bias4,
                                 float* __restrict__ h_cur,
                                 float* __restrict__ c_cur,
                                 float* __restrict__ q,
                                 const float* __restrict__ init,
                                 const float* __restrict__ bi,
                                 const float* __restrict__ bf_,
                                 const float* __restrict__ bc,
                                 const float* __restrict__ bo,
                                 const float* __restrict__ Wa) {
    int i = blockIdx.x * 256 + threadIdx.x;
    if (i < 512) {
        const float* bs[4] = {bi, bf_, bc, bo};
        bias4[i] = bs[i >> 7][i & 127];
    }
    if (i < BB * DD) {
        h_cur[i] = init[i];
        c_cur[i] = init[BB * DD + i];
        int b = i >> 7, d = i & 127;
        float acc = 0.0f;
        for (int k = 0; k < DD; ++k)
            acc = fmaf(init[b * DD + k], Wa[k * DD + d], acc);
        q[i] = acc;
    }
}

// ---------------------------------------------------------------------------
// HU = H @ Ua   (M = B*S = 4096, N = 128, K = 128), f16 WMMA, f32 accum
// grid: 256 blocks (one 16-row M-tile each) x 256 threads (8 waves = 8 N-tiles)
// ---------------------------------------------------------------------------
__global__ void hu_gemm_kernel(const _Float16* __restrict__ Ha,
                               const _Float16* __restrict__ Uasw,
                               float* __restrict__ HU) {
    int mt   = blockIdx.x;
    int wave = threadIdx.x >> 5;   // n-tile 0..7
    int lane = threadIdx.x & 31;
    v8f acc = {};
#pragma unroll
    for (int kb = 0; kb < 4; ++kb) {
        v16h a = *reinterpret_cast<const v16h*>(Ha   + (((long)mt * 4 + kb) * 32 + lane) * 16);
        v16h b = *reinterpret_cast<const v16h*>(Uasw + (((long)kb * 8 + wave) * 32 + lane) * 16);
        acc = __builtin_amdgcn_wmma_f32_16x16x32_f16(false, a, false, b,
                                                     (short)0, acc, false, false);
    }
    int n  = wave * 16 + (lane & 15);
    int hi = lane >> 4;
#pragma unroll
    for (int r = 0; r < 8; ++r) {
        int row = mt * 16 + r + 8 * hi;
        HU[(long)row * DD + n] = acc[r];
    }
}

// ---------------------------------------------------------------------------
// XW[t][b][col] = x[b,t,:] @ [Wi|Wf|Wc|Wo] + bias4   (M = B*T = 1024, N = 512)
// grid: 64 blocks x 256 threads; each wave handles 4 N-tiles
// ---------------------------------------------------------------------------
__global__ void xw_gemm_kernel(const _Float16* __restrict__ xA,
                               const _Float16* __restrict__ W4sw,
                               const float* __restrict__ bias4,
                               float* __restrict__ XW) {
    int mt   = blockIdx.x;
    int wave = threadIdx.x >> 5;
    int lane = threadIdx.x & 31;
    int hi   = lane >> 4;
#pragma unroll
    for (int i = 0; i < 4; ++i) {
        int nt = wave * 4 + i;     // 0..31
        v8f acc = {};
#pragma unroll
        for (int kb = 0; kb < 4; ++kb) {
            v16h a = *reinterpret_cast<const v16h*>(xA   + (((long)mt * 4 + kb) * 32 + lane) * 16);
            v16h b = *reinterpret_cast<const v16h*>(W4sw + (((long)kb * 32 + nt) * 32 + lane) * 16);
            acc = __builtin_amdgcn_wmma_f32_16x16x32_f16(false, a, false, b,
                                                         (short)0, acc, false, false);
        }
        int col = nt * 16 + (lane & 15);
        float bb = bias4[col];
#pragma unroll
        for (int r = 0; r < 8; ++r) {
            int row = mt * 16 + r + 8 * hi;   // row = b*T + t
            int b_  = row >> 7;
            int t_  = row & 127;
            XW[((long)t_ * BB + b_) * 512 + col] = acc[r] + bb;
        }
    }
}

// ---------------------------------------------------------------------------
// Attention step: per batch b, score_s = v . tanh(HU[b,s,:] + q[b,:]),
// softmax over S, context[b,:] = sum_s a_s * H[b,s,:]
// grid: 8 blocks x 512 threads; all global traffic as float4 (b128)
// ---------------------------------------------------------------------------
__global__ void att_kernel(const float* __restrict__ HU,
                           const float* __restrict__ H,
                           const float* __restrict__ q,
                           const float* __restrict__ v,
                           float* __restrict__ ctx) {
    int b   = blockIdx.x;
    int tid = threadIdx.x;                 // 0..511 == s
    __shared__ __align__(16) float  sq[DD];
    __shared__ __align__(16) float  sv[DD];
    __shared__ __align__(16) float  sc[SS];
    __shared__ __align__(16) float  red[SS];
    __shared__ __align__(16) float4 red4[SS];

    if (tid < DD) { sq[tid] = q[b * DD + tid]; sv[tid] = v[tid]; }
    __syncthreads();

    // score for s = tid, vectorized 4-wide
    const float4* hu4 = (const float4*)(HU + ((long)b * SS + tid) * DD);
    float score = 0.0f;
    for (int d4 = 0; d4 < DD / 4; ++d4) {
        float4 h4 = hu4[d4];
        float4 q4 = *(const float4*)&sq[d4 * 4];
        float4 v4 = *(const float4*)&sv[d4 * 4];
        score = fmaf(v4.x, tanhf(h4.x + q4.x), score);
        score = fmaf(v4.y, tanhf(h4.y + q4.y), score);
        score = fmaf(v4.z, tanhf(h4.z + q4.z), score);
        score = fmaf(v4.w, tanhf(h4.w + q4.w), score);
    }

    red[tid] = score;
    __syncthreads();
    for (int off = 256; off > 0; off >>= 1) {
        if (tid < off) red[tid] = fmaxf(red[tid], red[tid + off]);
        __syncthreads();
    }
    float mx = red[0];
    __syncthreads();
    float ex = __expf(score - mx);
    red[tid] = ex;
    __syncthreads();
    for (int off = 256; off > 0; off >>= 1) {
        if (tid < off) red[tid] += red[tid + off];
        __syncthreads();
    }
    float inv = 1.0f / red[0];
    __syncthreads();
    sc[tid] = ex * inv;
    __syncthreads();

    // context: thread -> (d4 = tid&31 owns d = 4*d4..4*d4+3, chunk = tid>>5),
    // 32 s-rows each; H reads are b128 and coalesce across the 32 threads/chunk
    int d4    = tid & 31;
    int chunk = tid >> 5;                  // 0..15
    const float* hb = H + ((long)b * SS + chunk * 32) * DD + d4 * 4;
    float4 part = {0.0f, 0.0f, 0.0f, 0.0f};
    for (int j = 0; j < 32; ++j) {
        float  w  = sc[chunk * 32 + j];
        float4 hh = *(const float4*)(hb + (long)j * DD);
        part.x = fmaf(w, hh.x, part.x);
        part.y = fmaf(w, hh.y, part.y);
        part.z = fmaf(w, hh.z, part.z);
        part.w = fmaf(w, hh.w, part.w);
    }
    red4[tid] = part;
    __syncthreads();
    for (int off = 256; off >= 32; off >>= 1) {
        if (tid < off) {
            float4 a = red4[tid], c = red4[tid + off];
            a.x += c.x; a.y += c.y; a.z += c.z; a.w += c.w;
            red4[tid] = a;
        }
        __syncthreads();
    }
    if (tid < 32)
        ((float4*)ctx)[b * 32 + tid] = red4[tid];
}

// ---------------------------------------------------------------------------
// Gate step: pre = h@[Ui|Uf|Uc|Uo|Wa] + ctx@[Ci|Cf|Cc|Co] (cols 0..511),
// then LSTM elementwise update + emit q_{t+1} (cols 512..639).
// 1 block x 256 threads (8 waves, 5 N-tiles each, N = 640)
// ---------------------------------------------------------------------------
__global__ void gate_kernel(const _Float16* __restrict__ U5sw,
                            const _Float16* __restrict__ C4sw,
                            const float* __restrict__ XW_t,
                            const float* __restrict__ ctx,
                            float* __restrict__ h_cur,
                            float* __restrict__ c_cur,
                            float* __restrict__ q,
                            float* __restrict__ out, int t) {
    __shared__ __align__(32) _Float16 a_h[4 * 512];
    __shared__ __align__(32) _Float16 a_c[4 * 512];
    __shared__ __align__(16) float pre[BB * 640];
    int tid = threadIdx.x;

    // Stage h (padded 16x128) and ctx into WMMA A-layout, f32 -> f16
    for (int i = tid; i < 2048; i += 256) {
        int e  = i & 15;
        int l  = (i >> 4) & 31;
        int kb = i >> 9;
        int m  = l & 15;
        int k  = kb * 32 + (e & 7) + ((e & 8) ? 16 : 0) + 8 * (l >> 4);
        _Float16 hv = (_Float16)0.0f, cv = (_Float16)0.0f;
        if (m < BB) {
            hv = (_Float16)h_cur[m * DD + k];
            cv = (_Float16)ctx[m * DD + k];
        }
        a_h[i] = hv;
        a_c[i] = cv;
    }
    __syncthreads();

    int wave = tid >> 5;
    int lane = tid & 31;
    int hi   = lane >> 4;
#pragma unroll
    for (int i = 0; i < 5; ++i) {
        int nt = wave * 5 + i;     // 0..39 (wave-uniform)
        v8f acc = {};
#pragma unroll
        for (int kb = 0; kb < 4; ++kb) {
            v16h a = *reinterpret_cast<const v16h*>(&a_h[kb * 512 + lane * 16]);
            v16h bm = *reinterpret_cast<const v16h*>(U5sw + (((long)kb * 40 + nt) * 32 + lane) * 16);
            acc = __builtin_amdgcn_wmma_f32_16x16x32_f16(false, a, false, bm,
                                                         (short)0, acc, false, false);
        }
        if (nt < 32) {   // ctx @ C-gates only covers cols 0..511
#pragma unroll
            for (int kb = 0; kb < 4; ++kb) {
                v16h a = *reinterpret_cast<const v16h*>(&a_c[kb * 512 + lane * 16]);
                v16h bm = *reinterpret_cast<const v16h*>(C4sw + (((long)kb * 32 + nt) * 32 + lane) * 16);
                acc = __builtin_amdgcn_wmma_f32_16x16x32_f16(false, a, false, bm,
                                                             (short)0, acc, false, false);
            }
        }
        int n = nt * 16 + (lane & 15);
#pragma unroll
        for (int r = 0; r < 8; ++r) {
            int m = r + 8 * hi;
            if (m < BB) pre[m * 640 + n] = acc[r];
        }
    }
    __syncthreads();

    // LSTM elementwise update + next-step q, 4-wide (b128 traffic)
    for (int i = tid; i < BB * DD / 4; i += 256) {
        int b  = i >> 5;          // batch
        int d4 = i & 31;          // group of 4 d's
        const float* pb = pre + b * 640 + d4 * 4;
        const float* xw = XW_t + b * 512 + d4 * 4;
        float4 xi = *(const float4*)(xw);
        float4 xf = *(const float4*)(xw + 128);
        float4 xg = *(const float4*)(xw + 256);
        float4 xo = *(const float4*)(xw + 384);
        float4 cp = ((const float4*)c_cur)[i];
        float4 cn, hn;
        {
            float ig = sigmoidf_(pb[0]   + xi.x);
            float fg = sigmoidf_(pb[128] + xf.x);
            float gg = tanhf    (pb[256] + xg.x);
            float og = sigmoidf_(pb[384] + xo.x);
            cn.x = fg * cp.x + ig * gg;  hn.x = og * tanhf(cn.x);
        }
        {
            float ig = sigmoidf_(pb[1]     + xi.y);
            float fg = sigmoidf_(pb[129]   + xf.y);
            float gg = tanhf    (pb[257]   + xg.y);
            float og = sigmoidf_(pb[385]   + xo.y);
            cn.y = fg * cp.y + ig * gg;  hn.y = og * tanhf(cn.y);
        }
        {
            float ig = sigmoidf_(pb[2]     + xi.z);
            float fg = sigmoidf_(pb[130]   + xf.z);
            float gg = tanhf    (pb[258]   + xg.z);
            float og = sigmoidf_(pb[386]   + xo.z);
            cn.z = fg * cp.z + ig * gg;  hn.z = og * tanhf(cn.z);
        }
        {
            float ig = sigmoidf_(pb[3]     + xi.w);
            float fg = sigmoidf_(pb[131]   + xf.w);
            float gg = tanhf    (pb[259]   + xg.w);
            float og = sigmoidf_(pb[387]   + xo.w);
            cn.w = fg * cp.w + ig * gg;  hn.w = og * tanhf(cn.w);
        }
        ((float4*)c_cur)[i] = cn;
        ((float4*)h_cur)[i] = hn;
        ((float4*)out)[((long)b * TT + t) * (DD / 4) + d4] = hn;
        float4 qv = *(const float4*)(pb + 512);
        ((float4*)q)[i] = qv;     // h_t @ Wa for next step
    }
}

// ---------------------------------------------------------------------------
extern "C" void kernel_launch(void* const* d_in, const int* in_sizes, int n_in,
                              void* d_out, int out_size, void* d_ws, size_t ws_size,
                              hipStream_t stream) {
    const float* x    = (const float*)d_in[0];
    const float* H    = (const float*)d_in[1];
    const float* init = (const float*)d_in[2];
    const float* Wa   = (const float*)d_in[3];
    const float* Ua   = (const float*)d_in[4];
    const float* v    = (const float*)d_in[5];
    const float* Wi   = (const float*)d_in[6];
    const float* Ui   = (const float*)d_in[7];
    const float* Ci   = (const float*)d_in[8];
    const float* bi   = (const float*)d_in[9];
    const float* Wf   = (const float*)d_in[10];
    const float* Uf   = (const float*)d_in[11];
    const float* Cf   = (const float*)d_in[12];
    const float* bf_  = (const float*)d_in[13];
    const float* Wc   = (const float*)d_in[14];
    const float* Uc   = (const float*)d_in[15];
    const float* Cc   = (const float*)d_in[16];
    const float* bc   = (const float*)d_in[17];
    const float* Wo   = (const float*)d_in[18];
    const float* Uo   = (const float*)d_in[19];
    const float* Co   = (const float*)d_in[20];
    const float* bo   = (const float*)d_in[21];
    float* out = (float*)d_out;

    // Carve workspace (256B-aligned chunks)
    char*  base = (char*)d_ws;
    size_t off  = 0;
    auto carve = [&](size_t bytes) -> char* {
        char* p = base + off;
        off += (bytes + 255) & ~(size_t)255;
        return p;
    };
    float*    HU    = (float*)carve((size_t)BB * SS * DD * 4);   // 2 MB
    float*    XW    = (float*)carve((size_t)TT * BB * 512 * 4);  // 2 MB
    float*    ctx   = (float*)carve(BB * DD * 4);
    float*    q     = (float*)carve(BB * DD * 4);
    float*    h_cur = (float*)carve(BB * DD * 4);
    float*    c_cur = (float*)carve(BB * DD * 4);
    float*    bias4 = (float*)carve(512 * 4);
    _Float16* Ha    = (_Float16*)carve((size_t)BB * SS * DD * 2);  // 1 MB
    _Float16* xA    = (_Float16*)carve((size_t)BB * TT * DD * 2);
    _Float16* Uasw  = (_Float16*)carve(4 * 8 * 32 * 16 * 2);
    _Float16* W4sw  = (_Float16*)carve(4 * 32 * 32 * 16 * 2);
    _Float16* U5sw  = (_Float16*)carve(4 * 40 * 32 * 16 * 2);
    _Float16* C4sw  = (_Float16*)carve(4 * 32 * 32 * 16 * 2);

    // --- precompute / swizzle phase (parallel) ---
    {
        int total = BB * SS * DD;   // 524288 = 256 M-tiles
        swizzleA_kernel<<<(total + 255) / 256, 256, 0, stream>>>(Ha, H, total);
    }
    {
        int total = BB * TT * DD;   // 131072 = 64 M-tiles
        swizzleA_kernel<<<(total + 255) / 256, 256, 0, stream>>>(xA, x, total);
    }
    swizzleB_kernel<<<(4 * 8 * 512 + 255) / 256, 256, 0, stream>>>(
        Uasw, Ua, Ua, Ua, Ua, Ua, 8);
    swizzleB_kernel<<<(4 * 32 * 512 + 255) / 256, 256, 0, stream>>>(
        W4sw, Wi, Wf, Wc, Wo, Wo, 32);
    swizzleB_kernel<<<(4 * 40 * 512 + 255) / 256, 256, 0, stream>>>(
        U5sw, Ui, Uf, Uc, Uo, Wa, 40);
    swizzleB_kernel<<<(4 * 32 * 512 + 255) / 256, 256, 0, stream>>>(
        C4sw, Ci, Cf, Cc, Co, Co, 32);
    misc_prep_kernel<<<8, 256, 0, stream>>>(bias4, h_cur, c_cur, q, init,
                                            bi, bf_, bc, bo, Wa);

    hu_gemm_kernel<<<256, 256, 0, stream>>>(Ha, Uasw, HU);
    xw_gemm_kernel<<<64, 256, 0, stream>>>(xA, W4sw, bias4, XW);

    // --- sequential recurrence over T ---
    for (int t = 0; t < TT; ++t) {
        att_kernel<<<BB, 512, 0, stream>>>(HU, H, q, v, ctx);
        gate_kernel<<<1, 256, 0, stream>>>(U5sw, C4sw, XW + (size_t)t * BB * 512,
                                           ctx, h_cur, c_cur, q, out, t);
    }
}
